// BiLSTM_CRF_38225208934541
// MI455X (gfx1250) — matrix operand order repaired and checked by reading, not compiled, not run on hardware
//
#include <hip/hip_runtime.h>
#include <hip/hip_bf16.h>

// ---------------------------------------------------------------------------
// BiLSTM-CRF loss for MI455X (gfx1250), wave32 + WMMA bf16.
// T=512, B=128, V=32000, E=256, H=256, G=4H=1024, K=32 tags.
// ---------------------------------------------------------------------------

#define TT 512
#define BB 128
#define EE 256
#define HH 256
#define GG 1024
#define KK 32

// sched_group_barrier masks
#define SGB_WMMA   0x008
#define SGB_VMEMRD 0x020
#define SGB_DSRD   0x100
#define SGB_MEMRD  (SGB_VMEMRD | SGB_DSRD)

typedef __attribute__((ext_vector_type(16))) __bf16 v16bf;
typedef __attribute__((ext_vector_type(8)))  __bf16 v8bf;
typedef __attribute__((ext_vector_type(8)))  float  v8f;

__device__ __forceinline__ unsigned short f2bf(float f) {
    unsigned u = __builtin_bit_cast(unsigned, f);
    unsigned r = u + 0x7fffu + ((u >> 16) & 1u);   // round-to-nearest-even
    return (unsigned short)(r >> 16);
}
__device__ __forceinline__ float bf2f(unsigned short h) {
    return __builtin_bit_cast(float, (unsigned)h << 16);
}
__device__ __forceinline__ float sigf(float x) { return 1.f / (1.f + __expf(-x)); }
__device__ __forceinline__ float tanhr(float x) {
    float e = __expf(-2.f * x);
    return (1.f - e) / (1.f + e);
}
__device__ __forceinline__ v16bf comb16(v8bf lo, v8bf hi) {
    v16bf r;
#pragma unroll
    for (int i = 0; i < 8; ++i) { r[i] = lo[i]; r[i + 8] = hi[i]; }
    return r;
}
__device__ __forceinline__ v8f wmma_bf16(v16bf a, v16bf b, v8f c) {
    return __builtin_amdgcn_wmma_f32_16x16x32_bf16(false, a, false, b, (short)0, c,
                                                   false, false);
}
// Opaque zero derived from the loop counter: makes weight addresses look
// loop-variant so LICM cannot hoist the streaming loads (which would exhaust
// VGPRs and spill to scratch), while the loads stay ordinary global_load_b128
// (relaxed scope, clause-able, partial s_wait_loadcnt).
__device__ __forceinline__ int opaque_zero(int t) {
    int z;
    asm("s_mul_i32 %0, %1, 0" : "=s"(z) : "s"(t));
    return z;
}

// ---------------------------------------------------------------------------
// f32 -> bf16 weight conversion
// ---------------------------------------------------------------------------
__global__ void cvt_kernel(const float* __restrict__ src,
                           unsigned short* __restrict__ dst, int n) {
    int i = blockIdx.x * 256 + threadIdx.x;
    if (i < n) dst[i] = f2bf(src[i]);
}

// ---------------------------------------------------------------------------
// Embedding gather -> bf16 (T,B,E)
// ---------------------------------------------------------------------------
__global__ void embed_kernel(const int* __restrict__ seqs,
                             const float* __restrict__ emb,
                             unsigned short* __restrict__ xbf) {
    size_t i = (size_t)blockIdx.x * 256 + threadIdx.x;   // T*B*E elements
    int e  = (int)(i & (EE - 1));
    int tb = (int)(i >> 8);
    int v  = seqs[tb];
    xbf[i] = f2bf(emb[(size_t)v * EE + e]);
}

// ---------------------------------------------------------------------------
// LSTM direction kernel. grid = (8 batch tiles, 2 directions), block = 512.
// Wave w owns hidden columns [16w, 16w+16); computes gate tiles i,f,g,o via
// WMMA over K=512 (x part from global bf16, h part from LDS). Weights are
// streamed from L2 every step (2 MB total, L2-resident); operand registers
// are double-buffered and a sched_group_barrier pipeline keeps ~2 K-chunks
// of loads in flight behind the WMMAs.
// ---------------------------------------------------------------------------
__global__ __launch_bounds__(512) void lstm_kernel(
    const unsigned short* __restrict__ xbf,
    const unsigned short* __restrict__ WihF, const unsigned short* __restrict__ WhhF,
    const unsigned short* __restrict__ WihB, const unsigned short* __restrict__ WhhB,
    const float* __restrict__ bihF, const float* __restrict__ bhhF,
    const float* __restrict__ bihB, const float* __restrict__ bhhB,
    const int* __restrict__ masks,
    unsigned short* __restrict__ hfOut, unsigned short* __restrict__ hbOut) {
    const int dir = blockIdx.y;
    const int b0  = blockIdx.x * 16;
    const int tid = threadIdx.x;
    const int w    = tid >> 5;
    const int lane = tid & 31;
    const int ln = lane & 15;   // N / row-in-tile index
    const int lh = lane >> 4;   // half-wave select
    const int hc0 = w * 16;

    const unsigned short* Wih = dir ? WihB : WihF;
    const unsigned short* Whh = dir ? WhhB : WhhF;
    const float* bih = dir ? bihB : bihF;
    const float* bhh = dir ? bhhB : bhhF;
    unsigned short* hout = dir ? hbOut : hfOut;

    __shared__ __align__(32) unsigned short hcur[16 * HH];   // 8 KB h tile
    for (int i = tid; i < 16 * HH; i += 512) hcur[i] = 0;
    __syncthreads();

    // Per-gate weight base pointers (constant per wave/lane; per-step K-chunk
    // offsets fold into the load instruction's immediate offset field).
    const unsigned short* wx[4];
    const unsigned short* wh[4];
    float bias[4];
#pragma unroll
    for (int g = 0; g < 4; ++g) {
        int col = g * HH + hc0 + ln;
        wx[g] = Wih + (size_t)col * EE + lh * 16;
        wh[g] = Whh + (size_t)col * HH + lh * 16;
        bias[g] = bih[col] + bhh[col];
        // Warm L2 for this wave's weight rows (global_prefetch_b8).
        __builtin_prefetch(wx[g], 0, 1);
        __builtin_prefetch(wh[g], 0, 1);
        __builtin_prefetch(wx[g] + 256, 0, 1);
        __builtin_prefetch(wh[g] + 256, 0, 1);
    }

    v8f c = {};       // cell state tile (C/D layout)
    v8f hprev = {};   // previous h for this wave's columns

    for (int t = 0; t < TT; ++t) {
        const int tt = dir ? (TT - 1 - t) : t;
        const int z  = opaque_zero(t);   // always 0; defeats LICM on weights
        v8f acc[4] = {{}, {}, {}, {}};

        const unsigned short* wxz[4];
        const unsigned short* whz[4];
#pragma unroll
        for (int g = 0; g < 4; ++g) {
            wxz[g] = wx[g] + z;
            whz[g] = wh[g] + z;
        }

        // ---- x contribution: K = 0..255, double-buffered pipeline ----
        const unsigned short* xrow = xbf + ((size_t)tt * BB + b0 + ln) * EE;
        v8bf xa[2][2];
        v8bf xb[2][4][2];
        {
            const v8bf* pa = (const v8bf*)(xrow + lh * 16);
            xa[0][0] = pa[0];
            xa[0][1] = pa[1];
#pragma unroll
            for (int g = 0; g < 4; ++g) {
                const v8bf* pb = (const v8bf*)(wxz[g]);
                xb[0][g][0] = pb[0];
                xb[0][g][1] = pb[1];
            }
        }
#pragma unroll
        for (int kk = 0; kk < 8; ++kk) {
            const int cur = kk & 1, nxt = cur ^ 1;
            if (kk < 7) {   // issue next chunk's loads before this chunk's WMMAs
                const v8bf* pa = (const v8bf*)(xrow + (kk + 1) * 32 + lh * 16);
                xa[nxt][0] = pa[0];
                xa[nxt][1] = pa[1];
#pragma unroll
                for (int g = 0; g < 4; ++g) {
                    const v8bf* pb = (const v8bf*)(wxz[g] + (kk + 1) * 32);
                    xb[nxt][g][0] = pb[0];
                    xb[nxt][g][1] = pb[1];
                }
            }
            v16bf a = comb16(xa[cur][0], xa[cur][1]);
#pragma unroll
            for (int g = 0; g < 4; ++g)
                acc[g] = wmma_bf16(a, comb16(xb[cur][g][0], xb[cur][g][1]), acc[g]);
        }

        // ---- h contribution: K = 0..255, A from LDS, B double-buffered ----
        const unsigned short* hrow = hcur + ln * HH;
        {
#pragma unroll
            for (int g = 0; g < 4; ++g) {
                const v8bf* pb = (const v8bf*)(whz[g]);
                xb[0][g][0] = pb[0];
                xb[0][g][1] = pb[1];
            }
        }
#pragma unroll
        for (int kk = 0; kk < 8; ++kk) {
            const int cur = kk & 1, nxt = cur ^ 1;
            if (kk < 7) {
#pragma unroll
                for (int g = 0; g < 4; ++g) {
                    const v8bf* pb = (const v8bf*)(whz[g] + (kk + 1) * 32);
                    xb[nxt][g][0] = pb[0];
                    xb[nxt][g][1] = pb[1];
                }
            }
            const v8bf* pa = (const v8bf*)(hrow + kk * 32 + lh * 16);   // ds_load_b128
            v16bf a = comb16(pa[0], pa[1]);
#pragma unroll
            for (int g = 0; g < 4; ++g)
                acc[g] = wmma_bf16(a, comb16(xb[cur][g][0], xb[cur][g][1]), acc[g]);
        }

        // ---- scheduling pipeline: keep ~2 K-chunks of loads in flight ----
        // Leading group: chunks 0-1 loads; then 16x { 4 WMMA ; next chunk's
        // ~10 memory reads }. Trailing unmatched groups are harmless.
        __builtin_amdgcn_sched_group_barrier(SGB_MEMRD, 20, 0);
#pragma unroll
        for (int i = 0; i < 16; ++i) {
            __builtin_amdgcn_sched_group_barrier(SGB_WMMA, 4, 0);
            __builtin_amdgcn_sched_group_barrier(SGB_MEMRD, 10, 0);
        }

        __syncthreads();   // all waves done reading hcur for this step

        // ---- elementwise LSTM cell update ----
#pragma unroll
        for (int r = 0; r < 8; ++r) {
            float iv = sigf(acc[0][r] + bias[0]);
            float fv = sigf(acc[1][r] + bias[1]);
            float gv = tanhr(acc[2][r] + bias[2]);
            float ov = sigf(acc[3][r] + bias[3]);
            float cn = fv * c[r] + iv * gv;
            float hn = ov * tanhr(cn);
            int M = r + (lh << 3);                 // row within 16-row tile
            int m = masks[tt * BB + b0 + M];
            float cs = m ? cn : c[r];
            float hs = m ? hn : hprev[r];
            c[r] = cs;
            hprev[r] = hs;
            hcur[M * HH + hc0 + ln] = f2bf(hs);
            float ho = m ? hn : 0.f;
            hout[((size_t)tt * BB + b0 + M) * HH + hc0 + ln] = f2bf(ho);
        }
        __syncthreads();   // new hcur visible to all waves
    }
}

// ---------------------------------------------------------------------------
// em = concat(hf,hb) @ Wout^T + bout.  M = T*B = 65536, N = 32, K = 512.
// One wave per (16-row M-tile, 16-col N-tile): 8192 wave tasks.
// ---------------------------------------------------------------------------
__global__ __launch_bounds__(256) void em_kernel(
    const unsigned short* __restrict__ hf, const unsigned short* __restrict__ hb,
    const unsigned short* __restrict__ WoutB, const float* __restrict__ bout,
    float* __restrict__ em) {
    int gw = blockIdx.x * 8 + (threadIdx.x >> 5);
    int lane = threadIdx.x & 31;
    int ln = lane & 15, lh = lane >> 4;
    int mtile = gw >> 1, ntile = gw & 1;
    int r0 = mtile * 16;
    int n0 = ntile * 16;

    v8f acc = {};
    const unsigned short* af = hf + (size_t)(r0 + ln) * HH;
    const unsigned short* ab = hb + (size_t)(r0 + ln) * HH;
#pragma unroll
    for (int kk = 0; kk < 16; ++kk) {
        const unsigned short* base = (kk < 8) ? (af + kk * 32) : (ab + (kk - 8) * 32);
        const v8bf* pa = (const v8bf*)(base + lh * 16);
        v16bf a = comb16(pa[0], pa[1]);
        const v16bf* pb = (const v16bf*)(WoutB +
            (size_t)(n0 + ln) * (2 * HH) + kk * 32 + lh * 16);
        acc = wmma_bf16(a, *pb, acc);
    }
    float bv = bout[n0 + ln];
#pragma unroll
    for (int r = 0; r < 8; ++r) {
        int M = r + (lh << 3);
        em[(size_t)(r0 + M) * KK + n0 + ln] = acc[r] + bv;
    }
}

// ---------------------------------------------------------------------------
// CRF numerator (gold-path score). One thread per batch element.
// ---------------------------------------------------------------------------
__global__ __launch_bounds__(128) void crf_num_kernel(
    const int* __restrict__ tags, const int* __restrict__ masks,
    const float* __restrict__ em, const float* __restrict__ start_trans,
    const float* __restrict__ end_trans, const float* __restrict__ trans,
    float* __restrict__ numer) {
    __shared__ float tr[KK * KK];
    for (int i = threadIdx.x; i < KK * KK; i += 128) tr[i] = trans[i];
    __syncthreads();
    int b = threadIdx.x;
    int tprev = tags[b];
    float score = start_trans[tprev] + em[(size_t)b * KK + tprev];
    int len = masks[b] ? 1 : 0;
    for (int t = 1; t < TT; ++t) {
        int m  = masks[t * BB + b];
        int tc = tags[t * BB + b];
        if (m) {
            score += tr[tprev * KK + tc] + em[((size_t)t * BB + b) * KK + tc];
            ++len;
        }
        tprev = tc;
    }
    int last = len - 1;
    score += end_trans[tags[last * BB + b]];
    numer[b] = score;
}

// ---------------------------------------------------------------------------
// CRF forward algorithm (logZ). One wave per batch element; lane = tag state.
// ---------------------------------------------------------------------------
__global__ __launch_bounds__(32) void crf_logz_kernel(
    const int* __restrict__ masks, const float* __restrict__ em,
    const float* __restrict__ start_trans, const float* __restrict__ end_trans,
    const float* __restrict__ trans, float* __restrict__ lz) {
    int b = blockIdx.x;
    int k = threadIdx.x;
    __shared__ float tr[KK * KK];
    for (int i = k; i < KK * KK; i += 32) tr[i] = trans[i];
    __syncthreads();

    float alpha = start_trans[k] + em[(size_t)b * KK + k];
    for (int t = 1; t < TT; ++t) {
        float e = em[((size_t)t * BB + b) * KK + k];
        float m = -3.4e38f;
#pragma unroll 8
        for (int j = 0; j < KK; ++j) {
            float aj = __shfl(alpha, j, 32);
            m = fmaxf(m, aj + tr[j * KK + k]);
        }
        float s = 0.f;
#pragma unroll 8
        for (int j = 0; j < KK; ++j) {
            float aj = __shfl(alpha, j, 32);
            s += __expf(aj + tr[j * KK + k] - m);
        }
        float nxt = m + __logf(s) + e;
        alpha = masks[t * BB + b] ? nxt : alpha;
    }
    float v = alpha + end_trans[k];
    float m = v;
#pragma unroll
    for (int o = 16; o > 0; o >>= 1) m = fmaxf(m, __shfl_xor(m, o, 32));
    float s = __expf(v - m);
#pragma unroll
    for (int o = 16; o > 0; o >>= 1) s += __shfl_xor(s, o, 32);
    if (k == 0) lz[b] = m + __logf(s);
}

// ---------------------------------------------------------------------------
// Final deterministic reduction: loss = mean(logZ - numerator)
// ---------------------------------------------------------------------------
__global__ __launch_bounds__(128) void loss_kernel(const float* __restrict__ numer,
                                                   const float* __restrict__ lz,
                                                   float* __restrict__ out) {
    __shared__ float sbuf[128];
    int i = threadIdx.x;
    sbuf[i] = lz[i] - numer[i];
    __syncthreads();
    for (int s = 64; s > 0; s >>= 1) {
        if (i < s) sbuf[i] += sbuf[i + s];
        __syncthreads();
    }
    if (i == 0) out[0] = sbuf[0] / (float)BB;
}

// ---------------------------------------------------------------------------
extern "C" void kernel_launch(void* const* d_in, const int* in_sizes, int n_in,
                              void* d_out, int out_size, void* d_ws, size_t ws_size,
                              hipStream_t stream) {
    const int*   seqs  = (const int*)d_in[0];
    const int*   tags  = (const int*)d_in[1];
    const int*   masks = (const int*)d_in[2];
    const float* emb   = (const float*)d_in[3];
    const float* WihF  = (const float*)d_in[4];
    const float* WhhF  = (const float*)d_in[5];
    const float* bihF  = (const float*)d_in[6];
    const float* bhhF  = (const float*)d_in[7];
    const float* WihB  = (const float*)d_in[8];
    const float* WhhB  = (const float*)d_in[9];
    const float* bihB  = (const float*)d_in[10];
    const float* bhhB  = (const float*)d_in[11];
    const float* Wout  = (const float*)d_in[12];
    const float* bout  = (const float*)d_in[13];
    const float* stT   = (const float*)d_in[14];
    const float* enT   = (const float*)d_in[15];
    const float* trans = (const float*)d_in[16];

    char* ws = (char*)d_ws;
    size_t off = 0;
    auto alloc = [&](size_t bytes) {
        size_t o = off;
        off = (off + bytes + 255) & ~(size_t)255;
        return o;
    };
    unsigned short* xbf   = (unsigned short*)(ws + alloc((size_t)TT * BB * EE * 2));
    unsigned short* hfB   = (unsigned short*)(ws + alloc((size_t)TT * BB * HH * 2));
    unsigned short* hbB   = (unsigned short*)(ws + alloc((size_t)TT * BB * HH * 2));
    unsigned short* wihfB = (unsigned short*)(ws + alloc((size_t)GG * EE * 2));
    unsigned short* whhfB = (unsigned short*)(ws + alloc((size_t)GG * HH * 2));
    unsigned short* wihbB = (unsigned short*)(ws + alloc((size_t)GG * EE * 2));
    unsigned short* whhbB = (unsigned short*)(ws + alloc((size_t)GG * HH * 2));
    unsigned short* woutB = (unsigned short*)(ws + alloc((size_t)KK * 2 * HH * 2));
    float*          emBuf = (float*)(ws + alloc((size_t)TT * BB * KK * 4));
    float*          numer = (float*)(ws + alloc((size_t)BB * 4));
    float*          lzBuf = (float*)(ws + alloc((size_t)BB * 4));

    // weight conversion f32 -> bf16
    cvt_kernel<<<(GG * EE + 255) / 256, 256, 0, stream>>>(WihF, wihfB, GG * EE);
    cvt_kernel<<<(GG * HH + 255) / 256, 256, 0, stream>>>(WhhF, whhfB, GG * HH);
    cvt_kernel<<<(GG * EE + 255) / 256, 256, 0, stream>>>(WihB, wihbB, GG * EE);
    cvt_kernel<<<(GG * HH + 255) / 256, 256, 0, stream>>>(WhhB, whhbB, GG * HH);
    cvt_kernel<<<(KK * 2 * HH + 255) / 256, 256, 0, stream>>>(Wout, woutB, KK * 2 * HH);

    // embedding gather
    embed_kernel<<<(TT * BB * EE) / 256, 256, 0, stream>>>(seqs, emb, xbf);

    // bidirectional LSTM (WMMA recurrence)
    lstm_kernel<<<dim3(BB / 16, 2), 512, 0, stream>>>(
        xbf, wihfB, whhfB, wihbB, whhbB, bihF, bhhF, bihB, bhhB, masks, hfB, hbB);

    // emission GEMM
    em_kernel<<<(TT * BB / 16) * 2 / 8, 256, 0, stream>>>(hfB, hbB, woutB, bout, emBuf);

    // CRF
    crf_num_kernel<<<1, 128, 0, stream>>>(tags, masks, emBuf, stT, enT, trans, numer);
    crf_logz_kernel<<<BB, 32, 0, stream>>>(masks, emBuf, stT, enT, trans, lzBuf);
    loss_kernel<<<1, 128, 0, stream>>>(numer, lzBuf, (float*)d_out);
}
